// EfficientSelfAttention2D_4303557231288
// MI455X (gfx1250) — matrix operand order repaired and checked by reading, not compiled, not run on hardware
//
#include <hip/hip_runtime.h>
#include <hip/hip_bf16.h>

#define BATCH 2
#define CCH   512
#define NTOK  4096      // 64*64
#define NHEAD 8
#define HD    64
#define GRP   32
#define CPG   16        // CCH/GRP
#define C3    1536
#define PLD   72        // padded LDS row stride (36 banks) for 64-elem bf16 rows

typedef __attribute__((ext_vector_type(16))) __bf16 v16bf;
typedef __attribute__((ext_vector_type(8)))  __bf16 v8bf;
typedef __attribute__((ext_vector_type(8)))  float  v8f;
typedef __attribute__((ext_vector_type(4)))  unsigned int u32x4;
typedef __attribute__((ext_vector_type(8)))  int i32x8;
typedef __attribute__((ext_vector_type(4)))  int i32x4;

#if defined(__HIP_DEVICE_COMPILE__) && __has_builtin(__builtin_amdgcn_tensor_load_to_lds)
#define HAVE_TDM 1
#else
#define HAVE_TDM 0
#endif

// ---- WMMA fragment helpers (bf16 16x16x32, f32 accum) ---------------------
// A-matrix per-lane layout (wave32): lane l, m = l&15, kb = 16*(l>>4):
//   elements e -> K = kb+e
__device__ __forceinline__ v16bf make_afrag(const __bf16* p) {
  v8bf lo = *(const v8bf*)p;
  v8bf hi = *(const v8bf*)(p + 8);
  return __builtin_shufflevector(lo, hi, 0,1,2,3,4,5,6,7,8,9,10,11,12,13,14,15);
}
// B-matrix per-lane layout: lane l, n = l&15, elements e -> K = 16*(l>>4)+e
__device__ __forceinline__ v16bf make_bfrag(const __bf16* p) {   // 32B-aligned
  return *(const v16bf*)p;
}
__device__ __forceinline__ v16bf make_bfrag16(const __bf16* p) { // 16B-aligned
  v8bf lo = *(const v8bf*)p;
  v8bf hi = *(const v8bf*)(p + 8);
  return __builtin_shufflevector(lo, hi, 0,1,2,3,4,5,6,7,8,9,10,11,12,13,14,15);
}
__device__ __forceinline__ v8f wmma_bf16(v16bf a, v16bf b, v8f c) {
  return __builtin_amdgcn_wmma_f32_16x16x32_bf16(false, a, false, b, (short)0, c,
                                                 false, false);
}

#if HAVE_TDM
// ---- TDM: 2D tile (tile_w x tile_h elems, 2-byte) global -> LDS -----------
// D# per CDNA5 ISA 8.3-8.6. pad=true inserts 4 DWORDs after every 32 DWORDs
// (rows are exactly 32 DW) -> 144B LDS row stride = PLD elements.
__device__ __forceinline__ void tdm_load_2d(const __bf16* lds, const __bf16* g,
                                            unsigned tile_w, unsigned tile_h,
                                            unsigned long long row_stride,
                                            unsigned tdim0, unsigned tdim1,
                                            bool pad) {
  u32x4 g0;
  i32x8 g1;
  i32x4 g2 = {0, 0, 0, 0}, g3 = {0, 0, 0, 0};
  const unsigned long long ga = (unsigned long long)(size_t)g;
  g0[0] = 1u;                                           // count=1, user mode
  g0[1] = (unsigned)(size_t)lds;                        // LDS byte offset
  g0[2] = (unsigned)ga;                                 // global_addr[31:0]
  g0[3] = (unsigned)((ga >> 32) & 0x01ffffffu) | (2u << 30);  // addr[56:32]|type=2
  unsigned w0 = (1u << 16);                             // data_size code 1 = 2B
  if (pad) w0 |= (1u << 20) | (4u << 22) | (3u << 25);  // pad_en, 32DW int., 4DW amt
  g1[0] = (int)w0;
  g1[1] = (int)((tdim0 & 0xffffu) << 16);               // tensor_dim0[15:0]
  g1[2] = (int)((tdim0 >> 16) | ((tdim1 & 0xffffu) << 16));
  g1[3] = (int)((tdim1 >> 16) | (tile_w << 16));        // tile_dim0
  g1[4] = (int)tile_h;                                  // tile_dim1 (tile_dim2=0)
  g1[5] = (int)(row_stride & 0xffffffffull);            // dim0_stride[31:0]
  g1[6] = (int)((row_stride >> 32) & 0xffffull);        // dim0_stride[47:32]
  g1[7] = 0;
#if __clang_major__ >= 23
  i32x8 g4 = {0, 0, 0, 0, 0, 0, 0, 0};
  __builtin_amdgcn_tensor_load_to_lds(g0, g1, g2, g3, g4, 0);
#else
  __builtin_amdgcn_tensor_load_to_lds(g0, g1, g2, g3, 0);
#endif
}
#endif

// ---- 1) GroupNorm statistics ----------------------------------------------
__global__ void __launch_bounds__(256)
gn_stats_kernel(const float* __restrict__ x, float* __restrict__ stats) {
  const int bg = blockIdx.x;                       // b*32 + g (groups contiguous)
  const float* px = x + (size_t)bg * CPG * NTOK;
  float s = 0.f, ss = 0.f;
  for (int i = threadIdx.x; i < CPG * NTOK; i += 256) {
    float v = px[i]; s += v; ss += v * v;
  }
  __shared__ float r1[256], r2[256];
  r1[threadIdx.x] = s; r2[threadIdx.x] = ss;
  __syncthreads();
  for (int o = 128; o > 0; o >>= 1) {
    if ((int)threadIdx.x < o) {
      r1[threadIdx.x] += r1[threadIdx.x + o];
      r2[threadIdx.x] += r2[threadIdx.x + o];
    }
    __syncthreads();
  }
  if (threadIdx.x == 0) {
    float mean = r1[0] * (1.f / (CPG * NTOK));
    float var  = r2[0] * (1.f / (CPG * NTOK)) - mean * mean;
    stats[bg * 2 + 0] = mean;
    stats[bg * 2 + 1] = rsqrtf(var + 1e-5f);
  }
}

// ---- 2) Apply GN + transpose to token-major (B,N,C) bf16, LDS-tiled -------
__global__ void __launch_bounds__(256)
gn_apply_kernel(const float* __restrict__ x, const float* __restrict__ gw,
                const float* __restrict__ gb, const float* __restrict__ stats,
                __bf16* __restrict__ xn) {
  __shared__ float t[64][65];
  const int b = blockIdx.z;
  const int n0 = blockIdx.x * 64, c0 = blockIdx.y * 64;
  const int tx = threadIdx.x & 63, ty = threadIdx.x >> 6;
  #pragma unroll
  for (int i = 0; i < 64; i += 4) {
    const int c = c0 + ty + i;
    const int g = c >> 4;
    const float mean = stats[(b * GRP + g) * 2 + 0];
    const float rstd = stats[(b * GRP + g) * 2 + 1];
    const float sc = rstd * gw[c];
    const float sh = gb[c] - mean * sc;
    t[ty + i][tx] = x[((size_t)(b * CCH + c)) * NTOK + n0 + tx] * sc + sh;
  }
  __syncthreads();
  #pragma unroll
  for (int i = 0; i < 64; i += 4)
    xn[((size_t)(b * NTOK + n0 + ty + i)) * CCH + c0 + tx] = (__bf16)t[tx][ty + i];
}

// ---- 3) Weights fp32 -> bf16 ----------------------------------------------
__global__ void __launch_bounds__(256)
convw_kernel(const float* __restrict__ qw, const float* __restrict__ pw,
             __bf16* __restrict__ qwb, __bf16* __restrict__ pwb) {
  const int i = blockIdx.x * 256 + threadIdx.x;
  if (i < C3 * CCH)  qwb[i] = (__bf16)qw[i];
  if (i < CCH * CCH) pwb[i] = (__bf16)pw[i];
}

// ---- 4) QKV GEMM: qkv[b,n,o] = sum_c xn[b,n,c] * qkv_w[o,c] + bias[o] -----
// Q pre-scaled by d^-0.5 * log2(e) so attention softmax runs in base-2.
__global__ void __launch_bounds__(256)
qkv_gemm_kernel(const __bf16* __restrict__ xn, const __bf16* __restrict__ wq,
                const float* __restrict__ bias, __bf16* __restrict__ qb,
                __bf16* __restrict__ kbuf, __bf16* __restrict__ vbuf) {
  const int lane = threadIdx.x & 31, wave = threadIdx.x >> 5;
  const int wm = wave >> 1, wn = wave & 1;         // 4 x 2 wave grid
  const int b = blockIdx.z;
  const int row0 = blockIdx.y * 128 + wm * 32;     // token rows
  const int col0 = blockIdx.x * 128 + wn * 64;     // output channels
  const int lm = lane & 15, lh = lane >> 4;
  const __bf16* A = xn + (size_t)b * NTOK * CCH;

  v8f acc[2][4];
  #pragma unroll
  for (int i = 0; i < 2; i++)
    #pragma unroll
    for (int j = 0; j < 4; j++)
      #pragma unroll
      for (int r = 0; r < 8; r++) acc[i][j][r] = 0.f;

  for (int c0 = 0; c0 < CCH; c0 += 32) {
    v16bf af[2], bf[4];
    #pragma unroll
    for (int i = 0; i < 2; i++)
      af[i] = make_afrag(A + (size_t)(row0 + i * 16 + lm) * CCH + c0 + (lh << 4));
    #pragma unroll
    for (int j = 0; j < 4; j++)
      bf[j] = make_bfrag(wq + (size_t)(col0 + j * 16 + lm) * CCH + c0 + (lh << 4));
    #pragma unroll
    for (int i = 0; i < 2; i++)
      #pragma unroll
      for (int j = 0; j < 4; j++)
        acc[i][j] = wmma_bf16(af[i], bf[j], acc[i][j]);
  }

  const float QSCALE = 0.125f * 1.44269504088896340736f;  // d^-0.5 * log2(e)
  #pragma unroll
  for (int i = 0; i < 2; i++) {
    #pragma unroll
    for (int j = 0; j < 4; j++) {
      const int o = col0 + j * 16 + lm;
      const float bs = bias[o];
      #pragma unroll
      for (int r = 0; r < 8; r++) {
        const int n = row0 + i * 16 + r + (lh << 3);
        const float v = acc[i][j][r] + bs;
        if (o < CCH) {                               // Q (b,h,n,d)
          const int h = o >> 6, d = o & 63;
          qb[(((size_t)(b * NHEAD + h)) * NTOK + n) * HD + d] = (__bf16)(v * QSCALE);
        } else if (o < 2 * CCH) {                    // K (b,h,n,d)
          const int oo = o - CCH, h = oo >> 6, d = oo & 63;
          kbuf[(((size_t)(b * NHEAD + h)) * NTOK + n) * HD + d] = (__bf16)v;
        } else {                                     // V natural (b,h,n,d)
          const int oo = o - 2 * CCH, h = oo >> 6, d = oo & 63;
          vbuf[(((size_t)(b * NHEAD + h)) * NTOK + n) * HD + d] = (__bf16)v;
        }
      }
    }
  }
}

// ---- 5) V transpose (b,h,n,d) -> (b,h,d,n), LDS-tiled ----------------------
__global__ void __launch_bounds__(256)
vtrans_kernel(const __bf16* __restrict__ v, __bf16* __restrict__ vt) {
  __shared__ __bf16 t[64][65];
  const int bh = blockIdx.y, n0 = blockIdx.x * 64;
  const __bf16* src = v + (size_t)bh * NTOK * HD;
  __bf16* dst = vt + (size_t)bh * HD * NTOK;
  const int tx = threadIdx.x & 63, ty = threadIdx.x >> 6;
  #pragma unroll
  for (int i = 0; i < 64; i += 4)
    t[ty + i][tx] = src[(size_t)(n0 + ty + i) * HD + tx];     // coalesced in d
  __syncthreads();
  #pragma unroll
  for (int i = 0; i < 64; i += 4)
    dst[(size_t)(ty + i) * NTOK + n0 + tx] = t[tx][ty + i];   // coalesced in n
}

// ---- 6) Flash attention: softmax2(Q K^T) V per (b, head), 64-key tiles ----
// TDM path: wave 0 double-buffers K / V^T 64x64 tiles into LDS (padded rows),
// tracked with TENSORcnt; all 4 waves consume shared tiles (4x less L2 read).
__global__ void __launch_bounds__(128)
attn_kernel(const __bf16* __restrict__ qb, const __bf16* __restrict__ kbuf,
            const __bf16* __restrict__ vtb, __bf16* __restrict__ ao) {
  const int lane = threadIdx.x & 31, wave = threadIdx.x >> 5;  // 4 waves
  const int lm = lane & 15, lh = lane >> 4;
  const int bh = blockIdx.y;
  const int q0 = blockIdx.x * 64 + wave * 16;      // 16 query rows per wave
  const __bf16* Q  = qb   + (size_t)bh * NTOK * HD;
  const __bf16* K  = kbuf + (size_t)bh * NTOK * HD;
  const __bf16* VT = vtb  + (size_t)bh * HD * NTOK;

  __shared__ __bf16 Pl[4][16][PLD];                // wave-private P transpose tile
#if HAVE_TDM
  __shared__ __bf16 Kl[2][64][PLD];
  __shared__ __bf16 Vl[2][64][PLD];
#endif

  v16bf qf[2];
  #pragma unroll
  for (int kc = 0; kc < 2; kc++)
    qf[kc] = make_afrag(Q + (size_t)(q0 + lm) * HD + kc * 32 + (lh << 4));

  v8f O[4];
  float mrun[8], lrun[8];
  #pragma unroll
  for (int f = 0; f < 4; f++)
    #pragma unroll
    for (int r = 0; r < 8; r++) O[f][r] = 0.f;
  #pragma unroll
  for (int r = 0; r < 8; r++) { mrun[r] = -3.0e38f; lrun[r] = 0.f; }

#if HAVE_TDM
  if (wave == 0) {
    tdm_load_2d(&Kl[0][0][0], K, 64, 64, HD, HD, NTOK, true);
    tdm_load_2d(&Vl[0][0][0], VT, 64, 64, NTOK, NTOK, HD, true);
  }
#endif

  const int NT = NTOK / 64;
  for (int t = 0; t < NT; t++) {
    const int key0 = t * 64;
#if HAVE_TDM
    if (wave == 0) {
      if (t + 1 < NT) {
        tdm_load_2d(&Kl[(t + 1) & 1][0][0], K + (size_t)(key0 + 64) * HD,
                    64, 64, HD, HD, NTOK, true);
        tdm_load_2d(&Vl[(t + 1) & 1][0][0], VT + (key0 + 64),
                    64, 64, NTOK, NTOK, HD, true);
        __builtin_amdgcn_s_wait_tensorcnt(2);      // tile t landed
      } else {
        __builtin_amdgcn_s_wait_tensorcnt(0);
      }
    }
    __syncthreads();
    const __bf16* Kt = &Kl[t & 1][0][0];
    const __bf16* Vt = &Vl[t & 1][0][0];
#endif
    // S = Q K^T for 16 q-rows x 64 keys (scores already in log2 domain)
    v8f s[4];
    #pragma unroll
    for (int j = 0; j < 4; j++) {
      #pragma unroll
      for (int r = 0; r < 8; r++) s[j][r] = 0.f;
      #pragma unroll
      for (int kc = 0; kc < 2; kc++) {
#if HAVE_TDM
        v16bf kf = make_bfrag16(Kt + (j * 16 + lm) * PLD + kc * 32 + (lh << 4));
#else
        v16bf kf = make_bfrag(K + (size_t)(key0 + j * 16 + lm) * HD +
                              kc * 32 + (lh << 4));
#endif
        s[j] = wmma_bf16(qf[kc], kf, s[j]);
      }
    }
    // Online softmax in base-2. C-frag row r = matrix row r + 8*lh, matching
    // the per-lane mrun/lrun state, so updates are lane-local after the
    // 16-lane shuffle reductions.
    #pragma unroll
    for (int r = 0; r < 8; r++) {
      float mx = fmaxf(fmaxf(s[0][r], s[1][r]), fmaxf(s[2][r], s[3][r]));
      #pragma unroll
      for (int off = 1; off < 16; off <<= 1)
        mx = fmaxf(mx, __shfl_xor(mx, off, 32));
      const float mnew = fmaxf(mrun[r], mx);
      const float corr = exp2f(mrun[r] - mnew);
      float p[4];
      #pragma unroll
      for (int j = 0; j < 4; j++) p[j] = exp2f(s[j][r] - mnew);
      float rs = (p[0] + p[1]) + (p[2] + p[3]);
      #pragma unroll
      for (int off = 1; off < 16; off <<= 1)
        rs += __shfl_xor(rs, off, 32);
      lrun[r] = lrun[r] * corr + rs;
      mrun[r] = mnew;
      #pragma unroll
      for (int f = 0; f < 4; f++) O[f][r] *= corr;
      const int prow = r + (lh << 3);
      #pragma unroll
      for (int j = 0; j < 4; j++)
        Pl[wave][prow][j * 16 + lm] = (__bf16)p[j];
    }
    asm volatile("" ::: "memory");                 // order wave-local LDS st->ld
    v16bf pa[2];
    #pragma unroll
    for (int kc2 = 0; kc2 < 2; kc2++)
      pa[kc2] = make_afrag(&Pl[wave][lm][kc2 * 32 + (lh << 4)]);
    asm volatile("" ::: "memory");
    // O += P V  (V^T tile: rows = d, per-lane contiguous along keys)
    #pragma unroll
    for (int f = 0; f < 4; f++) {
      #pragma unroll
      for (int kc2 = 0; kc2 < 2; kc2++) {
#if HAVE_TDM
        v16bf vf = make_bfrag16(Vt + (f * 16 + lm) * PLD + kc2 * 32 + (lh << 4));
#else
        v16bf vf = make_bfrag(VT + (size_t)(f * 16 + lm) * NTOK + key0 +
                              kc2 * 32 + (lh << 4));
#endif
        O[f] = wmma_bf16(pa[kc2], vf, O[f]);
      }
    }
#if HAVE_TDM
    __syncthreads();                               // buffer reusable for t+2
#endif
  }

  const int b = bh >> 3, h = bh & 7;
  #pragma unroll
  for (int r = 0; r < 8; r++) {
    const float inv = 1.f / lrun[r];
    const int n = q0 + r + (lh << 3);
    #pragma unroll
    for (int f = 0; f < 4; f++) {
      const int d = f * 16 + lm;
      ao[((size_t)(b * NTOK + n)) * CCH + h * 64 + d] = (__bf16)(O[f][r] * inv);
    }
  }
}

// ---- 7) Proj GEMM + bias + residual, channel-major output -----------------
// Swapped orientation: A = proj_w (M=out-ch, K=c), B = ao (token-major (n,c)
// is exactly the B-operand layout). C-frag columns = tokens -> coalesced f32
// residual reads and output stores.
__global__ void __launch_bounds__(256)
proj_gemm_kernel(const __bf16* __restrict__ ao, const __bf16* __restrict__ wp,
                 const float* __restrict__ bias, const float* __restrict__ x,
                 float* __restrict__ out) {
  const int lane = threadIdx.x & 31, wave = threadIdx.x >> 5;
  const int wm = wave >> 1, wn = wave & 1;
  const int b = blockIdx.z;
  const int row0 = blockIdx.y * 128 + wm * 32;     // output channels
  const int col0 = blockIdx.x * 128 + wn * 64;     // tokens
  const int lm = lane & 15, lh = lane >> 4;
  const __bf16* Bm = ao + (size_t)b * NTOK * CCH;

  v8f acc[2][4];
  #pragma unroll
  for (int i = 0; i < 2; i++)
    #pragma unroll
    for (int j = 0; j < 4; j++)
      #pragma unroll
      for (int r = 0; r < 8; r++) acc[i][j][r] = 0.f;

  for (int c0 = 0; c0 < CCH; c0 += 32) {
    v16bf af[2], bf[4];
    #pragma unroll
    for (int i = 0; i < 2; i++)
      af[i] = make_afrag(wp + (size_t)(row0 + i * 16 + lm) * CCH + c0 + (lh << 4));
    #pragma unroll
    for (int j = 0; j < 4; j++)
      bf[j] = make_bfrag(Bm + (size_t)(col0 + j * 16 + lm) * CCH + c0 + (lh << 4));
    #pragma unroll
    for (int i = 0; i < 2; i++)
      #pragma unroll
      for (int j = 0; j < 4; j++)
        acc[i][j] = wmma_bf16(af[i], bf[j], acc[i][j]);
  }

  #pragma unroll
  for (int i = 0; i < 2; i++) {
    #pragma unroll
    for (int r = 0; r < 8; r++) {
      const int o = row0 + i * 16 + r + (lh << 3);
      const float bs = bias[o];
      #pragma unroll
      for (int j = 0; j < 4; j++) {
        const int n = col0 + j * 16 + lm;
        const size_t idx = ((size_t)(b * CCH + o)) * NTOK + n;
        out[idx] = x[idx] + bs + acc[i][j][r];
      }
    }
  }
}

// ---- host launch -----------------------------------------------------------
extern "C" void kernel_launch(void* const* d_in, const int* in_sizes, int n_in,
                              void* d_out, int out_size, void* d_ws, size_t ws_size,
                              hipStream_t stream) {
  (void)in_sizes; (void)n_in; (void)out_size; (void)ws_size;
  const float* x      = (const float*)d_in[0];
  const float* gn_w   = (const float*)d_in[1];
  const float* gn_b   = (const float*)d_in[2];
  const float* qkv_w  = (const float*)d_in[3];
  const float* qkv_b  = (const float*)d_in[4];
  const float* proj_w = (const float*)d_in[5];
  const float* proj_b = (const float*)d_in[6];
  float* out = (float*)d_out;

  char* ws = (char*)d_ws;
  size_t off = 0;
  float*  stats = (float*)(ws + off);  off += 1024;
  __bf16* xn    = (__bf16*)(ws + off); off += (size_t)BATCH * NTOK * CCH * 2;
  __bf16* qwb   = (__bf16*)(ws + off); off += (size_t)C3 * CCH * 2;
  __bf16* pwb   = (__bf16*)(ws + off); off += (size_t)CCH * CCH * 2;
  __bf16* qb    = (__bf16*)(ws + off); off += (size_t)BATCH * NHEAD * NTOK * HD * 2;
  __bf16* kbuf  = (__bf16*)(ws + off); off += (size_t)BATCH * NHEAD * NTOK * HD * 2;
  __bf16* vbuf  = (__bf16*)(ws + off); off += (size_t)BATCH * NHEAD * NTOK * HD * 2;
  __bf16* vtb   = (__bf16*)(ws + off); off += (size_t)BATCH * NHEAD * HD * NTOK * 2;
  __bf16* ao    = (__bf16*)(ws + off); off += (size_t)BATCH * NTOK * CCH * 2;

  gn_stats_kernel<<<BATCH * GRP, 256, 0, stream>>>(x, stats);
  gn_apply_kernel<<<dim3(NTOK / 64, CCH / 64, BATCH), 256, 0, stream>>>(
      x, gn_w, gn_b, stats, xn);
  convw_kernel<<<(C3 * CCH + 255) / 256, 256, 0, stream>>>(qkv_w, proj_w, qwb, pwb);
  qkv_gemm_kernel<<<dim3(C3 / 128, NTOK / 128, BATCH), 256, 0, stream>>>(
      xn, qwb, qkv_b, qb, kbuf, vbuf);
  vtrans_kernel<<<dim3(NTOK / 64, BATCH * NHEAD), 256, 0, stream>>>(vbuf, vtb);
  attn_kernel<<<dim3(NTOK / 64, BATCH * NHEAD), 128, 0, stream>>>(qb, kbuf, vtb, ao);
  proj_gemm_kernel<<<dim3(NTOK / 128, CCH / 128, BATCH), 256, 0, stream>>>(
      ao, pwb, proj_b, x, out);
}
